// TemporalGNN_64879775973436
// MI455X (gfx1250) — compile-verified, matrix operand order checked by reading, not run on hardware
//
#include <hip/hip_runtime.h>
#include <hip/hip_fp16.h>

typedef _Float16 h16;
typedef __attribute__((ext_vector_type(16))) _Float16 v16h;
typedef __attribute__((ext_vector_type(8)))  float    v8f;
typedef __attribute__((ext_vector_type(4)))  float    v4f;

#define NN   100000
#define NE   1600000
#define NR   8
#define EPSV 1e-5f

// ---------------------------------------------------------------------------
// zero a float buffer
__global__ void k_zero(float* __restrict__ p, long n) {
    long i = (long)blockIdx.x * blockDim.x + threadIdx.x;
    if (i < n) p[i] = 0.0f;
}

// ---------------------------------------------------------------------------
// x0h[n][k] = f16( bn32( emb[n_id[n]][k] ) )   (N x 32, row-major f16)
__global__ void k_embed_bn32(const int* __restrict__ n_id,
                             const float* __restrict__ emb,
                             const float* __restrict__ g, const float* __restrict__ b,
                             const float* __restrict__ m, const float* __restrict__ v,
                             h16* __restrict__ x0h) {
    unsigned i = blockIdx.x * blockDim.x + threadIdx.x;
    if (i >= (unsigned)NN * 32u) return;
    unsigned n = i >> 5;
    unsigned k = i & 31u;
    float x = emb[(size_t)n_id[n] * 32 + k];
    float s = g[k] * rsqrtf(v[k] + EPSV);
    x = (x - m[k]) * s + b[k];
    x0h[i] = (h16)x;
}

// ---------------------------------------------------------------------------
// Pack weights into per-lane WMMA B-fragments (f16).
// Layout: wpk[(((r*4 + ct)*nkb + kb)*32 + lane)*16 + i]
//   col = ct*16 + (lane&15); kh = lane>>4;
//   k   = kb*32 + (i<8 ? kh*8+i : 16 + kh*8 + (i-8))
// r < NR -> W[r] ([K,64] row-major); r == NR -> root ([K,64]).
__global__ void k_pack(const float* __restrict__ W, const float* __restrict__ root,
                       int K, h16* __restrict__ wpk) {
    int nkb = K >> 5;
    int tid = blockIdx.x * blockDim.x + threadIdx.x;
    int total = (NR + 1) * 4 * nkb * 32;
    if (tid >= total) return;
    int lane = tid & 31;
    int t = tid >> 5;
    int kb = t % nkb; t /= nkb;
    int ct = t % 4;   t /= 4;
    int r  = t;
    int col = ct * 16 + (lane & 15);
    int kh  = lane >> 4;
    const float* src = (r < NR) ? (W + (size_t)r * K * 64) : root;
    h16* dst = wpk + (size_t)tid * 16;
    #pragma unroll
    for (int i = 0; i < 16; ++i) {
        int k = kb * 32 + ((i < 8) ? (kh * 8 + i) : (16 + kh * 8 + (i - 8)));
        dst[i] = (h16)src[(size_t)k * 64 + col];
    }
}

// ---------------------------------------------------------------------------
// H[r][rowtile..][64] = X(f16) @ W_r(f16)  via v_wmma_f32_16x16x32_f16
// grid: (NN/16, NR+1), block: 128 (4 waves; wave = 16x16 output col tile)
// K is a template parameter: straight-line body, immediate load offsets.
template <int K>
__global__ __launch_bounds__(128)
void k_gemm_wmma(const h16* __restrict__ X, const h16* __restrict__ wpk,
                 float* __restrict__ H) {
    constexpr int NKB = K / 32;
    unsigned rowbase = blockIdx.x * 16u;
    unsigned r    = blockIdx.y;              // 0..NR  (NR == root)
    unsigned ct   = threadIdx.x >> 5;        // col tile 0..3
    unsigned lane = threadIdx.x & 31u;
    unsigned mrow = lane & 15u;
    unsigned kh   = lane >> 4;

    // base pointers; all kb variation folds into immediate offsets
    const h16* ap = X + (size_t)(rowbase + mrow) * K + kh * 8u;
    const h16* bp = wpk + ((size_t)(r * 4u + ct) * NKB * 32u + lane) * 16u;

    v8f acc = {};
    #pragma unroll
    for (int kb = 0; kb < NKB; ++kb) {
        v16h a, b;
        *reinterpret_cast<v4f*>(&a)       = *reinterpret_cast<const v4f*>(ap + kb * 32);
        *(reinterpret_cast<v4f*>(&a) + 1) = *reinterpret_cast<const v4f*>(ap + kb * 32 + 16);
        *reinterpret_cast<v4f*>(&b)       = *reinterpret_cast<const v4f*>(bp + kb * 512);
        *(reinterpret_cast<v4f*>(&b) + 1) = *reinterpret_cast<const v4f*>(bp + kb * 512 + 8);
        acc = __builtin_amdgcn_wmma_f32_16x16x32_f16(false, a, false, b,
                                                     (short)0, acc, false, false);
    }
    // D layout: vgpr g: lanes 0-15 -> row g, lanes 16-31 -> row g+8; N = lane&15
    float* out = H + ((size_t)r * NN + rowbase) * 64 + ct * 16u + (lane & 15u);
    unsigned roff = (lane >> 4) * 8u;
    #pragma unroll
    for (int g = 0; g < 8; ++g)
        out[(size_t)(g + roff) * 64] = acc[g];
}

// ---------------------------------------------------------------------------
// deg[dst*NR + etype] += 1   (shared by both layers)
__global__ void k_deg(const int* __restrict__ dst, const int* __restrict__ et,
                      float* __restrict__ deg) {
    unsigned e = blockIdx.x * blockDim.x + threadIdx.x;
    if (e >= NE) return;
    atomicAdd(deg + (size_t)dst[e] * NR + et[e], 1.0f);
}

// ---------------------------------------------------------------------------
// agg[dst][c] += H[etype][src][c] / max(deg[dst,etype],1)
// 64 consecutive threads = one edge -> broadcast index loads, coalesced H reads
__global__ void k_scatter(const int* __restrict__ src, const int* __restrict__ dst,
                          const int* __restrict__ et,
                          const float* __restrict__ H, const float* __restrict__ deg,
                          float* __restrict__ agg) {
    size_t i = (size_t)blockIdx.x * blockDim.x + threadIdx.x;
    if (i >= (size_t)NE * 64) return;
    unsigned e = (unsigned)(i >> 6);
    unsigned c = (unsigned)(i & 63u);
    int d = dst[e], s = src[e], t = et[e];
    float dg = deg[(size_t)d * NR + t];
    float norm = 1.0f / fmaxf(dg, 1.0f);
    float msg = H[((size_t)t * NN + s) * 64 + c] * norm;
    atomicAdd(agg + (size_t)d * 64 + c, msg);
}

// ---------------------------------------------------------------------------
// x1h = f16( bn64( relu( agg + H[root] + bias ) ) )
__global__ void k_combine_relu_bn(const float* __restrict__ agg, const float* __restrict__ H,
                                  const float* __restrict__ bias,
                                  const float* __restrict__ g, const float* __restrict__ b,
                                  const float* __restrict__ m, const float* __restrict__ v,
                                  h16* __restrict__ xh) {
    size_t i = (size_t)blockIdx.x * blockDim.x + threadIdx.x;
    if (i >= (size_t)NN * 64) return;
    unsigned n = (unsigned)(i >> 6);
    unsigned c = (unsigned)(i & 63u);
    float x = agg[i] + H[((size_t)NR * NN + n) * 64 + c] + bias[c];
    x = fmaxf(x, 0.0f);
    float s = g[c] * rsqrtf(v[c] + EPSV);
    x = (x - m[c]) * s + b[c];
    xh[i] = (h16)x;
}

// ---------------------------------------------------------------------------
// out = bn64b( agg + H[root] + bias )   (f32 output)
__global__ void k_final_bn(const float* __restrict__ agg, const float* __restrict__ H,
                           const float* __restrict__ bias,
                           const float* __restrict__ g, const float* __restrict__ b,
                           const float* __restrict__ m, const float* __restrict__ v,
                           float* __restrict__ out) {
    size_t i = (size_t)blockIdx.x * blockDim.x + threadIdx.x;
    if (i >= (size_t)NN * 64) return;
    unsigned n = (unsigned)(i >> 6);
    unsigned c = (unsigned)(i & 63u);
    float x = agg[i] + H[((size_t)NR * NN + n) * 64 + c] + bias[c];
    float s = g[c] * rsqrtf(v[c] + EPSV);
    out[i] = (x - m[c]) * s + b[c];
}

// ---------------------------------------------------------------------------
static inline unsigned gblk(long n, int b) { return (unsigned)((n + b - 1) / b); }

extern "C" void kernel_launch(void* const* d_in, const int* in_sizes, int n_in,
                              void* d_out, int out_size, void* d_ws, size_t ws_size,
                              hipStream_t stream) {
    const int*   n_id  = (const int*)d_in[0];
    const int*   eidx  = (const int*)d_in[1];
    const int*   srcp  = eidx;            // edge_index[0]
    const int*   dstp  = eidx + NE;       // edge_index[1]
    const int*   et    = (const int*)d_in[2];
    const float* emb   = (const float*)d_in[3];
    const float* W1    = (const float*)d_in[4];
    const float* root1 = (const float*)d_in[5];
    const float* b1    = (const float*)d_in[6];
    const float* W2    = (const float*)d_in[7];
    const float* root2 = (const float*)d_in[8];
    const float* b2    = (const float*)d_in[9];
    const float* bn32g = (const float*)d_in[10];
    const float* bn32b = (const float*)d_in[11];
    const float* bn32m = (const float*)d_in[12];
    const float* bn32v = (const float*)d_in[13];
    const float* bn64g = (const float*)d_in[14];
    const float* bn64b = (const float*)d_in[15];
    const float* bn64m = (const float*)d_in[16];
    const float* bn64v = (const float*)d_in[17];
    const float* bnBg  = (const float*)d_in[18];
    const float* bnBb  = (const float*)d_in[19];
    const float* bnBm  = (const float*)d_in[20];
    const float* bnBv  = (const float*)d_in[21];
    float* out = (float*)d_out;

    // ---- workspace carve-out (256B aligned) ----
    char* p = (char*)d_ws;
    auto take = [&](size_t bytes) -> char* {
        char* r = p;
        p += (bytes + 255) & ~(size_t)255;
        return r;
    };
    h16*   x0h  = (h16*)  take((size_t)NN * 32 * sizeof(h16));
    float* H    = (float*)take((size_t)(NR + 1) * NN * 64 * sizeof(float));
    float* deg  = (float*)take((size_t)NN * NR * sizeof(float));
    float* agg  = (float*)take((size_t)NN * 64 * sizeof(float));
    h16*   x1h  = (h16*)  take((size_t)NN * 64 * sizeof(h16));
    h16*   wpk1 = (h16*)  take((size_t)(NR + 1) * 4 * 1 * 32 * 16 * sizeof(h16));
    h16*   wpk2 = (h16*)  take((size_t)(NR + 1) * 4 * 2 * 32 * 16 * sizeof(h16));

    const int B = 256;

    // init + weight packing + activations
    k_zero<<<gblk((long)NN * NR, B), B, 0, stream>>>(deg, (long)NN * NR);
    k_zero<<<gblk((long)NN * 64, B), B, 0, stream>>>(agg, (long)NN * 64);
    k_pack<<<gblk((NR + 1) * 4 * 1 * 32, B), B, 0, stream>>>(W1, root1, 32, wpk1);
    k_pack<<<gblk((NR + 1) * 4 * 2 * 32, B), B, 0, stream>>>(W2, root2, 64, wpk2);
    k_embed_bn32<<<gblk((long)NN * 32, B), B, 0, stream>>>(n_id, emb,
                                                           bn32g, bn32b, bn32m, bn32v, x0h);
    k_deg<<<gblk(NE, B), B, 0, stream>>>(dstp, et, deg);

    // ---- layer 1 ----
    k_gemm_wmma<32><<<dim3(NN / 16, NR + 1), 128, 0, stream>>>(x0h, wpk1, H);
    k_scatter<<<gblk((long)NE * 64, B), B, 0, stream>>>(srcp, dstp, et, H, deg, agg);
    k_combine_relu_bn<<<gblk((long)NN * 64, B), B, 0, stream>>>(agg, H, b1,
                                                                bn64g, bn64b, bn64m, bn64v, x1h);

    // ---- layer 2 ----
    k_zero<<<gblk((long)NN * 64, B), B, 0, stream>>>(agg, (long)NN * 64);
    k_gemm_wmma<64><<<dim3(NN / 16, NR + 1), 128, 0, stream>>>(x1h, wpk2, H);
    k_scatter<<<gblk((long)NE * 64, B), B, 0, stream>>>(srcp, dstp, et, H, deg, agg);
    k_final_bn<<<gblk((long)NN * 64, B), B, 0, stream>>>(agg, H, b2,
                                                         bnBg, bnBb, bnBm, bnBv, out);
}
